// CanonCausalMultiheadAttn_11218454577171
// MI455X (gfx1250) — matrix operand order repaired and checked by reading, not compiled, not run on hardware
//
#include <hip/hip_runtime.h>
#include <hip/hip_bf16.h>

// ---------------- problem constants ----------------
#define D_MODEL   2048
#define N_HEADS   16
#define N_KV      4
#define D_HEAD    128
#define TOTAL_QKV 3072
#define BATCH     2
#define SEQ       2048
#define MROWS     (BATCH * SEQ)        // 4096
#define Q_DIM     (N_HEADS * D_HEAD)   // 2048
#define KV_DIM    (N_KV * D_HEAD)      // 512
#define INV_SQRT_DHEAD 0.08838834764831845f

typedef _Float16 v4h  __attribute__((ext_vector_type(4)));
typedef _Float16 v8h  __attribute__((ext_vector_type(8)));
typedef _Float16 v16h __attribute__((ext_vector_type(16)));
typedef float    v8f  __attribute__((ext_vector_type(8)));
typedef int      v4i  __attribute__((ext_vector_type(4)));

// ---- optional CDNA5 async global->LDS copy (guarded; safe if absent) ----
#if defined(__has_builtin)
#if __has_builtin(__builtin_amdgcn_global_load_async_to_lds_b128) && \
    __has_builtin(__builtin_amdgcn_s_wait_asynccnt)
#define HAVE_ASYNC_COPY 1
#endif
#endif
#ifndef HAVE_ASYNC_COPY
#define HAVE_ASYNC_COPY 0
#endif

#if HAVE_ASYNC_COPY
typedef __attribute__((address_space(1))) v4i gv4i;  // global <4 x i32>
typedef __attribute__((address_space(3))) v4i lv4i;  // LDS    <4 x i32>
__device__ __forceinline__ void async_copy16B(const _Float16* g, _Float16* l) {
  __builtin_amdgcn_global_load_async_to_lds_b128((gv4i*)g, (lv4i*)l, 0, 0);
}
__device__ __forceinline__ void async_wait0() {
  __builtin_amdgcn_s_wait_asynccnt(0);
}
#else
__device__ __forceinline__ void async_copy16B(const _Float16* g, _Float16* l) {
  *(v8h*)l = *(const v8h*)g;
}
__device__ __forceinline__ void async_wait0() {}
#endif

__device__ __forceinline__ v16h pack16(v8h lo, v8h hi) {
  return __builtin_shufflevector(lo, hi, 0, 1, 2, 3, 4, 5, 6, 7,
                                         8, 9, 10, 11, 12, 13, 14, 15);
}

__device__ __forceinline__ v8f wmma_16x16x32(v16h a, v16h b, v8f c) {
  // D = A(16x32 f16) * B(32x16 f16) + C(16x16 f32)
  return __builtin_amdgcn_wmma_f32_16x16x32_f16(false, a, false, b,
                                                (short)0, c, false, false);
}

// ---------------------------------------------------------------------------
// fp32 -> f16 convert, 4 elements per thread (all sizes divisible by 4)
// ---------------------------------------------------------------------------
__global__ void cvt_f16x4_kernel(const float* __restrict__ src,
                                 _Float16* __restrict__ dst, long n4) {
  long i = (long)blockIdx.x * blockDim.x + threadIdx.x;
  if (i < n4) {
    float4 v = ((const float4*)src)[i];
    v4h o = {(_Float16)v.x, (_Float16)v.y, (_Float16)v.z, (_Float16)v.w};
    ((v4h*)dst)[i] = o;
  }
}

// Pack w_q | w_k | w_v column-wise into one [D_MODEL, TOTAL_QKV] f16 matrix.
__global__ void pack_wqkv_kernel(const float* __restrict__ wq,
                                 const float* __restrict__ wk,
                                 const float* __restrict__ wv,
                                 _Float16* __restrict__ W) {
  long i = (long)blockIdx.x * blockDim.x + threadIdx.x;
  if (i >= (long)D_MODEL * TOTAL_QKV) return;
  int n = (int)(i % TOTAL_QKV);
  int k = (int)(i / TOTAL_QKV);
  float v;
  if (n < Q_DIM)               v = wq[(size_t)k * Q_DIM + n];
  else if (n < Q_DIM + KV_DIM) v = wk[(size_t)k * KV_DIM + (n - Q_DIM)];
  else                         v = wv[(size_t)k * KV_DIM + (n - Q_DIM - KV_DIM)];
  W[i] = (_Float16)v;
}

// ---------------------------------------------------------------------------
// WMMA GEMM: C[M,N] = A[M,K] f16 @ B[K,N] f16  (row-major, compile-time
// shapes; OutT selects f32 or f16 epilogue).
// Block tile 128x128, K-tile 64, 256 threads = 8 waves (4Mx2N), wave = 32x64.
// Double-buffered LDS: async-DMA prefetch of tile i+1 overlaps WMMA on tile i.
// LDS row strides padded to 72 halves (36 dwords) -> conflict-free frag reads.
// ---------------------------------------------------------------------------
#define APAD 72
template <int M, int N, int K, typename OutT>
__global__ __launch_bounds__(256) void gemm_f16_kernel(
    const _Float16* __restrict__ A, const _Float16* __restrict__ B,
    OutT* __restrict__ C) {
  __shared__ alignas(16) _Float16 As[2][128][APAD];   // [buf][m][k] (64 used)
  __shared__ alignas(16) _Float16 BsT[2][128][APAD];  // [buf][n][k] (64 used)

  const int t    = threadIdx.x;
  const int lane = t & 31;
  const int wid  = t >> 5;
  const int wm   = wid >> 1;   // 0..3
  const int wn   = wid & 1;    // 0..1
  const int l15  = lane & 15;
  const int half = lane >> 4;  // 0 or 1
  const int m0   = blockIdx.y * 128;
  const int n0   = blockIdx.x * 128;

  auto load_tile = [&](int kt, int buf) {
    // A tile 128x64: async DMA straight into LDS, 2 threads/row x 32 halves
    {
      int r  = t >> 1;
      int cs = (t & 1) << 5;
      const _Float16* src = A + (size_t)(m0 + r) * K + kt + cs;
#pragma unroll
      for (int j = 0; j < 4; ++j)
        async_copy16B(src + 8 * j, &As[buf][r][cs + 8 * j]);
    }
    // B tile 64x128 -> BsT[n][k] (transposed via scalar LDS stores)
    {
      int k  = t >> 2;            // 0..63
      int nn = (t & 3) << 5;      // 0,32,64,96
      const v8h* src = (const v8h*)(B + (size_t)(kt + k) * N + n0 + nn);
#pragma unroll
      for (int j = 0; j < 4; ++j) {
        v8h b = src[j];
#pragma unroll
        for (int e = 0; e < 8; ++e) BsT[buf][nn + 8 * j + e][k] = b[e];
      }
    }
  };

  v8f acc[2][4];
#pragma unroll
  for (int mf = 0; mf < 2; ++mf)
#pragma unroll
    for (int nf = 0; nf < 4; ++nf)
#pragma unroll
      for (int i = 0; i < 8; ++i) acc[mf][nf][i] = 0.0f;

  load_tile(0, 0);
  async_wait0();
  __syncthreads();

  for (int kt = 0; kt < K; kt += 64) {
    const int buf = (kt >> 6) & 1;
    if (kt + 64 < K) load_tile(kt + 64, buf ^ 1);  // prefetch next tile

#pragma unroll
    for (int kk = 0; kk < 2; ++kk) {
      v16h af[2];
      {
        int kb = kk * 32 + (half << 3);
#pragma unroll
        for (int mf = 0; mf < 2; ++mf) {
          int row = wm * 32 + mf * 16 + l15;
          af[mf] = pack16(*(const v8h*)&As[buf][row][kb],
                          *(const v8h*)&As[buf][row][kb + 16]);
        }
      }
      v16h bf[4];
      {
        int kb2 = kk * 32 + (half << 3);
#pragma unroll
        for (int nf = 0; nf < 4; ++nf) {
          int col = wn * 64 + nf * 16 + l15;
          bf[nf] = pack16(*(const v8h*)&BsT[buf][col][kb2],
                          *(const v8h*)&BsT[buf][col][kb2 + 16]);
        }
      }
#pragma unroll
      for (int mf = 0; mf < 2; ++mf)
#pragma unroll
        for (int nf = 0; nf < 4; ++nf)
          acc[mf][nf] = wmma_16x16x32(af[mf], bf[nf], acc[mf][nf]);
    }
    async_wait0();     // prefetch async ops complete before barrier
    __syncthreads();   // waits DScnt too -> transposed stores visible
  }

  // --- store C: compile-time N -> immediate offsets ---
#pragma unroll
  for (int mf = 0; mf < 2; ++mf) {
    int row0 = m0 + wm * 32 + mf * 16 + half * 8;
#pragma unroll
    for (int nf = 0; nf < 4; ++nf) {
      int col = n0 + wn * 64 + nf * 16 + l15;
      OutT* cp = C + (size_t)row0 * N + col;
#pragma unroll
      for (int i = 0; i < 8; ++i) cp[i * N] = (OutT)acc[mf][nf][i];
    }
  }
}

// ---------------------------------------------------------------------------
// Depthwise causal conv1d (K=4) + residual on f16 QKV, fp32 math, then split:
//   Qh [B,H,S,128] (pre-scaled by 1/sqrt(128)), Kh/Vh [B,KVH,S,128]
// ---------------------------------------------------------------------------
__global__ void conv_split_kernel(const _Float16* __restrict__ qkv,
                                  const float* __restrict__ cw,
                                  _Float16* __restrict__ Qh,
                                  _Float16* __restrict__ Kh,
                                  _Float16* __restrict__ Vh) {
  long idx = (long)blockIdx.x * blockDim.x + threadIdx.x;
  if (idx >= (long)MROWS * TOTAL_QKV) return;
  int c  = (int)(idx % TOTAL_QKV);
  long bs = idx / TOTAL_QKV;
  int s  = (int)(bs % SEQ);
  int b  = (int)(bs / SEQ);

  float w0 = cw[c * 4 + 0], w1 = cw[c * 4 + 1];
  float w2 = cw[c * 4 + 2], w3 = cw[c * 4 + 3];

  float x = (float)qkv[idx];
  float acc = x + x * w3;  // residual + k=3 tap (self)
  if (s >= 1) acc += (float)qkv[idx - 1L * TOTAL_QKV] * w2;
  if (s >= 2) acc += (float)qkv[idx - 2L * TOTAL_QKV] * w1;
  if (s >= 3) acc += (float)qkv[idx - 3L * TOTAL_QKV] * w0;

  if (c < Q_DIM) {
    int h = c >> 7, d = c & 127;
    Qh[(((size_t)b * N_HEADS + h) * SEQ + s) * D_HEAD + d] =
        (_Float16)(acc * INV_SQRT_DHEAD);
  } else if (c < Q_DIM + KV_DIM) {
    int kvh = (c - Q_DIM) >> 7, d = c & 127;
    Kh[(((size_t)b * N_KV + kvh) * SEQ + s) * D_HEAD + d] = (_Float16)acc;
  } else {
    int kvh = (c - Q_DIM - KV_DIM) >> 7, d = c & 127;
    Vh[(((size_t)b * N_KV + kvh) * SEQ + s) * D_HEAD + d] = (_Float16)acc;
  }
}

// ---------------------------------------------------------------------------
// Flash-style causal GQA attention.
// Grid: BATCH * N_HEADS * (SEQ/128); block 256 = 8 waves; wave owns 16 q-rows.
// Key/value tiles of 64 in LDS; online softmax; WMMA for QK^T and PV.
// Output: Att [B, S, H, 128] f16.
// ---------------------------------------------------------------------------
#define KSPAD 136   // 68 dwords, gcd(68,64)=4 -> conflict-free row reads
#define VTPAD 72
__global__ __launch_bounds__(256) void attn_kernel(
    const _Float16* __restrict__ Qh, const _Float16* __restrict__ Kh,
    const _Float16* __restrict__ Vh, _Float16* __restrict__ Att) {
  __shared__ alignas(16) _Float16 Ks[64][KSPAD];     // [key][d] (128 used)
  __shared__ alignas(16) _Float16 VT[128][VTPAD];    // [d][key] (64 used)
  __shared__ alignas(16) _Float16 Ps[8][16][VTPAD];  // per-wave P [row][key]

  const int t    = threadIdx.x;
  const int lane = t & 31;
  const int wid  = t >> 5;
  const int l15  = lane & 15;
  const int half = lane >> 4;

  const int blk = blockIdx.x;
  const int qc  = blk & 15;          // q chunk (128 rows)
  const int h   = (blk >> 4) & 15;
  const int b   = blk >> 8;
  const int kvh = h >> 2;            // GQA: 4 query heads per kv head
  const int q0  = qc * 128;
  const int qw  = q0 + wid * 16;     // this wave's first q row

  // --- Q fragments, resident in registers: 4 K-steps over d=128 ---
  const _Float16* qptr =
      Qh + (((size_t)b * N_HEADS + h) * SEQ + qw + l15) * D_HEAD;
  const int kbA = half << 3;
  v16h qf[4];
#pragma unroll
  for (int kk = 0; kk < 4; ++kk)
    qf[kk] = pack16(*(const v8h*)(qptr + kk * 32 + kbA),
                    *(const v8h*)(qptr + kk * 32 + kbA + 16));

  v8f acc_o[8];
  float m_run[8], l_run[8];
#pragma unroll
  for (int nf = 0; nf < 8; ++nf)
#pragma unroll
    for (int i = 0; i < 8; ++i) acc_o[nf][i] = 0.0f;
#pragma unroll
  for (int i = 0; i < 8; ++i) { m_run[i] = -1e30f; l_run[i] = 0.0f; }

  const _Float16* kg = Kh + ((size_t)b * N_KV + kvh) * SEQ * D_HEAD;
  const _Float16* vg = Vh + ((size_t)b * N_KV + kvh) * SEQ * D_HEAD;

  const int ntiles = (q0 + 128) >> 6;  // 64-key tiles up to block's max q row
  for (int it = 0; it < ntiles; ++it) {
    const int kt = it * 64;
    // --- K tile 64x128: async copy straight into LDS ---
    {
      int k  = t >> 2;           // 0..63
      int dd = (t & 3) << 5;     // 0,32,64,96
      const _Float16* src = kg + (size_t)(kt + k) * D_HEAD + dd;
#pragma unroll
      for (int j = 0; j < 4; ++j)
        async_copy16B(src + 8 * j, &Ks[k][dd + 8 * j]);
    }
    // --- V tile 64x128 -> VT[d][key] (transposed) ---
    {
      int k  = t >> 2;
      int dd = (t & 3) << 5;
      const v8h* src = (const v8h*)(vg + (size_t)(kt + k) * D_HEAD + dd);
#pragma unroll
      for (int j = 0; j < 4; ++j) {
        v8h v = src[j];
#pragma unroll
        for (int e = 0; e < 8; ++e) VT[dd + 8 * j + e][k] = v[e];
      }
    }
    async_wait0();
    __syncthreads();

    // --- scores S = Q @ K^T (four 16-key fragments) ---
    v8f sc[4];
#pragma unroll
    for (int nf = 0; nf < 4; ++nf)
#pragma unroll
      for (int i = 0; i < 8; ++i) sc[nf][i] = 0.0f;
#pragma unroll
    for (int kk = 0; kk < 4; ++kk) {
      int d0 = kk * 32 + (half << 3);
#pragma unroll
      for (int nf = 0; nf < 4; ++nf) {
        int key = nf * 16 + l15;
        v16h bfv = pack16(*(const v8h*)&Ks[key][d0],
                          *(const v8h*)&Ks[key][d0 + 16]);
        sc[nf] = wmma_16x16x32(qf[kk], bfv, sc[nf]);
      }
    }

    // --- causal mask + online softmax ---
    float mnew[8], psum[8];
#pragma unroll
    for (int i = 0; i < 8; ++i) {
      int qrow = qw + half * 8 + i;
      float m = -1e30f;
#pragma unroll
      for (int nf = 0; nf < 4; ++nf) {
        int kcol = kt + nf * 16 + l15;
        float s = (kcol > qrow) ? -1e30f : sc[nf][i];
        sc[nf][i] = s;
        m = fmaxf(m, s);
      }
      mnew[i] = m;
    }
#pragma unroll
    for (int xm = 1; xm < 16; xm <<= 1)
#pragma unroll
      for (int i = 0; i < 8; ++i)
        mnew[i] = fmaxf(mnew[i], __shfl_xor(mnew[i], xm, 32));
#pragma unroll
    for (int i = 0; i < 8; ++i) {
      float mt    = fmaxf(m_run[i], mnew[i]);
      float scale = __expf(m_run[i] - mt);
      float ps    = 0.0f;
#pragma unroll
      for (int nf = 0; nf < 4; ++nf) {
        float p = __expf(sc[nf][i] - mt);
        sc[nf][i] = p;
        ps += p;
      }
      psum[i]  = ps;
      m_run[i] = mt;
      l_run[i] *= scale;
#pragma unroll
      for (int nf = 0; nf < 8; ++nf) acc_o[nf][i] *= scale;
    }
#pragma unroll
    for (int xm = 1; xm < 16; xm <<= 1)
#pragma unroll
      for (int i = 0; i < 8; ++i) psum[i] += __shfl_xor(psum[i], xm, 32);
#pragma unroll
    for (int i = 0; i < 8; ++i) l_run[i] += psum[i];

    // --- P: C-layout -> LDS -> A-layout ---
#pragma unroll
    for (int nf = 0; nf < 4; ++nf)
#pragma unroll
      for (int i = 0; i < 8; ++i)
        Ps[wid][half * 8 + i][nf * 16 + l15] = (_Float16)sc[nf][i];
    __syncthreads();

    {
      int kb = half << 3;
      v16h pf0 = pack16(*(const v8h*)&Ps[wid][l15][kb],
                        *(const v8h*)&Ps[wid][l15][kb + 16]);
      v16h pf1 = pack16(*(const v8h*)&Ps[wid][l15][32 + kb],
                        *(const v8h*)&Ps[wid][l15][32 + kb + 16]);
      int kb2 = half << 3;
#pragma unroll
      for (int nfd = 0; nfd < 8; ++nfd) {
        int dcol = nfd * 16 + l15;
        v16h bf0 = pack16(*(const v8h*)&VT[dcol][kb2],
                          *(const v8h*)&VT[dcol][kb2 + 16]);
        v16h bf1 = pack16(*(const v8h*)&VT[dcol][32 + kb2],
                          *(const v8h*)&VT[dcol][32 + kb2 + 16]);
        acc_o[nfd] = wmma_16x16x32(pf0, bf0, acc_o[nfd]);
        acc_o[nfd] = wmma_16x16x32(pf1, bf1, acc_o[nfd]);
      }
    }
    __syncthreads();
  }

  // --- epilogue: normalize and store attended [B,S,H,128] f16 ---
#pragma unroll
  for (int nfd = 0; nfd < 8; ++nfd) {
    int d = nfd * 16 + l15;
#pragma unroll
    for (int i = 0; i < 8; ++i) {
      int qrow = qw + half * 8 + i;
      float o = acc_o[nfd][i] / l_run[i];
      Att[(((size_t)b * SEQ + qrow) * N_HEADS + h) * D_HEAD + d] = (_Float16)o;
    }
  }
}

// ---------------------------------------------------------------------------
extern "C" void kernel_launch(void* const* d_in, const int* in_sizes, int n_in,
                              void* d_out, int out_size, void* d_ws, size_t ws_size,
                              hipStream_t stream) {
  const float* hidden = (const float*)d_in[0];
  const float* w_q    = (const float*)d_in[1];
  const float* w_k    = (const float*)d_in[2];
  const float* w_v    = (const float*)d_in[3];
  const float* w_o    = (const float*)d_in[4];
  const float* conv_w = (const float*)d_in[5];
  float* out = (float*)d_out;

  char* ws = (char*)d_ws;
  size_t off = 0;
  auto take = [&](size_t bytes) {
    char* p = ws + off;
    off = (off + bytes + 255) & ~(size_t)255;
    return p;
  };
  _Float16* Xh   = (_Float16*)take((size_t)MROWS * D_MODEL * 2);
  _Float16* Wqkv = (_Float16*)take((size_t)D_MODEL * TOTAL_QKV * 2);
  _Float16* Woh  = (_Float16*)take((size_t)Q_DIM * D_MODEL * 2);
  _Float16* QKVh = (_Float16*)take((size_t)MROWS * TOTAL_QKV * 2);
  _Float16* Qh   = (_Float16*)take((size_t)MROWS * Q_DIM * 2);
  _Float16* Kh   = (_Float16*)take((size_t)MROWS * KV_DIM * 2);
  _Float16* Vh   = (_Float16*)take((size_t)MROWS * KV_DIM * 2);
  _Float16* Att  = (_Float16*)take((size_t)MROWS * Q_DIM * 2);

  {
    long n4 = (long)MROWS * D_MODEL / 4;
    cvt_f16x4_kernel<<<(int)((n4 + 255) / 256), 256, 0, stream>>>(hidden, Xh, n4);
  }
  {
    long n = (long)D_MODEL * TOTAL_QKV;
    pack_wqkv_kernel<<<(int)((n + 255) / 256), 256, 0, stream>>>(w_q, w_k, w_v, Wqkv);
  }
  {
    long n4 = (long)Q_DIM * D_MODEL / 4;
    cvt_f16x4_kernel<<<(int)((n4 + 255) / 256), 256, 0, stream>>>(w_o, Woh, n4);
  }
  // QKV projection: [4096,2048] @ [2048,3072] -> f16
  gemm_f16_kernel<MROWS, TOTAL_QKV, D_MODEL, _Float16>
      <<<dim3(TOTAL_QKV / 128, MROWS / 128), 256, 0, stream>>>(Xh, Wqkv, QKVh);
  {
    long n = (long)MROWS * TOTAL_QKV;
    conv_split_kernel<<<(int)((n + 255) / 256), 256, 0, stream>>>(QKVh, conv_w,
                                                                  Qh, Kh, Vh);
  }
  attn_kernel<<<BATCH * N_HEADS * (SEQ / 128), 256, 0, stream>>>(Qh, Kh, Vh, Att);
  // Output projection: [4096,2048] @ [2048,2048] -> d_out fp32
  gemm_f16_kernel<MROWS, D_MODEL, Q_DIM, float>
      <<<dim3(D_MODEL / 128, MROWS / 128), 256, 0, stream>>>(Att, Woh, out);
}
